// GAT_82248623718918
// MI455X (gfx1250) — matrix operand order repaired and checked by reading, not compiled, not run on hardware
//
#include <hip/hip_runtime.h>
#include <math.h>

// ---------------------------------------------------------------------------
// GAT 2-layer pipeline for gfx1250 (MI455X).
// Dense projections: v_wmma_f32_16x16x32_bf16 (wave32 WMMA), with the weight
// panel staged into LDS once per block via GLOBAL_LOAD_ASYNC_TO_LDS_B128
// (ASYNCcnt DMA) and consumed through clauses of ds_load_b128.
// Edge softmax/aggregation: f32 global atomics (L2-resident accumulators).
// ---------------------------------------------------------------------------

typedef __attribute__((ext_vector_type(16))) __bf16        v16bf;
typedef __attribute__((ext_vector_type(8)))  float         v8f;
typedef __attribute__((ext_vector_type(4)))  unsigned int  v4u;
typedef __attribute__((ext_vector_type(8)))  unsigned int  v8u;

__device__ __forceinline__ unsigned short f2bf(float f) {
    unsigned int u = __float_as_uint(f);
    u += 0x7fffu + ((u >> 16) & 1u);          // round-to-nearest-even
    return (unsigned short)(u >> 16);
}

// ------------------------------- utility kernels ---------------------------

__global__ void k_fill_f32(float* __restrict__ p, float v, long long n) {
    long long i = (long long)blockIdx.x * blockDim.x + threadIdx.x;
    if (i < n) p[i] = v;
}

__global__ void k_cvt_bf16(const float* __restrict__ in,
                           unsigned short* __restrict__ out, long long n) {
    long long i = (long long)blockIdx.x * blockDim.x + threadIdx.x;
    if (i < n) out[i] = f2bf(in[i]);
}

// W [K, Ncols] row-major f32  ->  Wt [NcolsPad, K] bf16 (zero pad cols)
__global__ void k_transpose_cvt(const float* __restrict__ W,
                                unsigned short* __restrict__ Wt,
                                int K, int Ncols, int NcolsPad) {
    int idx = blockIdx.x * blockDim.x + threadIdx.x;
    int total = NcolsPad * K;
    if (idx >= total) return;
    int n = idx / K;
    int k = idx - n * K;
    float v = (n < Ncols) ? W[(long long)k * Ncols + n] : 0.0f;
    Wt[(long long)n * K + k] = f2bf(v);
}

// ------------------------------- WMMA GEMM ---------------------------------
// C[rows, ccols] = A[rows,128](bf16) @ Bt[NT*16,128](bf16, col-major panels)
// Block: 256 threads = 8 waves; each wave computes 16 rows x NT*16 cols.
// Fragment layouts follow CDNA5 ISA 7.12.2 (16-bit A 16x32 / B 32x16, wave32).
template <int NT>
__global__ void k_gemm_bf16_wmma(const unsigned short* __restrict__ A,
                                 const unsigned short* __restrict__ Bt,
                                 float* __restrict__ C, int rows, int ccols) {
    constexpr int LDSS = 136;                 // 128 + 8 ushort pad (272B rows)
    __shared__ __attribute__((aligned(16))) unsigned short smem[NT * 16 * LDSS];

    // ---- stage weight panel (NT*16 x 128 bf16) into LDS via async DMA ----
    // 16B chunks; chunk c covers panel row n = c/16, cols k = (c%16)*8..+7.
    // Trip count is uniform across the block, so EXEC stays all-1s.
    #pragma unroll
    for (int c = threadIdx.x; c < NT * 256; c += 256) {
        int n = c >> 4;
        int k = (c & 15) << 3;
        unsigned lds_addr = (unsigned)(unsigned long long)&smem[n * LDSS + k];
        unsigned goff = (unsigned)((n * 128 + k) * 2);   // byte offset in Bt
        asm volatile("global_load_async_to_lds_b128 %0, %1, %2"
                     :: "v"(lds_addr), "v"(goff), "s"(Bt)
                     : "memory");
    }
    asm volatile("s_wait_asynccnt 0x0" ::: "memory");
    __syncthreads();

    const int lane = threadIdx.x & 31;
    const int wave = threadIdx.x >> 5;
    const int row0 = blockIdx.x * 128 + wave * 16;
    if (row0 >= rows) return;                 // wave-uniform: EXEC stays all-1s
    const int lr = lane & 15;                 // row (A) / col (B) within tile
    const int kh = lane >> 4;                 // which K-half this lane holds

    v8f acc[NT] = {};

    int r = row0 + lr;
    if (r >= rows) r = rows - 1;              // clamp (N%16==0 so never taken)
    const unsigned short* arow = A + (long long)r * 128;

    #pragma unroll
    for (int k0 = 0; k0 < 128; k0 += 32) {
        // ---- A fragment: lane holds K = k0+kh*8+[0..7] and k0+16+kh*8+[0..7]
        const unsigned short* ap = arow + k0 + kh * 8;
        if (k0 + 32 < 128) __builtin_prefetch((const void*)(ap + 32), 0, 0);
        v4u a0 = *(const v4u*)ap;
        v4u a1 = *(const v4u*)(ap + 16);
        v8u au;
        au[0] = a0[0]; au[1] = a0[1]; au[2] = a0[2]; au[3] = a0[3];
        au[4] = a1[0]; au[5] = a1[1]; au[6] = a1[2]; au[7] = a1[3];
        v16bf afrag = __builtin_bit_cast(v16bf, au);

        // ---- load ALL B fragments for this K-step into distinct registers
        // so the ds_load_b128s issue as one clause and waits stagger instead
        // of a full s_wait_dscnt 0 before every WMMA.
        v16bf bfrag[NT];
        #pragma unroll
        for (int t = 0; t < NT; ++t) {
            // B fragment from LDS: col n=t*16+lr, K = k0+kh*8+[0..7] and k0+16+kh*8+[0..7]
            const unsigned short* bp = &smem[(t * 16 + lr) * LDSS
                                             + k0 + kh * 8];
            v4u b0 = *(const v4u*)bp;
            v4u b1 = *(const v4u*)(bp + 16);
            v8u bu;
            bu[0] = b0[0]; bu[1] = b0[1]; bu[2] = b0[2]; bu[3] = b0[3];
            bu[4] = b1[0]; bu[5] = b1[1]; bu[6] = b1[2]; bu[7] = b1[3];
            bfrag[t] = __builtin_bit_cast(v16bf, bu);
        }

        #pragma unroll
        for (int t = 0; t < NT; ++t) {
            acc[t] = __builtin_amdgcn_wmma_f32_16x16x32_bf16(
                         false, afrag, false, bfrag[t],
                         (short)0, acc[t], false, false);
        }
    }

    // ---- store: VGPR j holds row (row0 + j + kh*8), col t*16+lr
    const int rb = row0 + (kh ? 8 : 0);
    #pragma unroll
    for (int t = 0; t < NT; ++t) {
        int c = t * 16 + lr;
        if (c < ccols) {
            #pragma unroll
            for (int j = 0; j < 8; ++j) {
                int rr = rb + j;
                if (rr < rows) C[(long long)rr * ccols + c] = acc[t][j];
            }
        }
    }
}

// ------------------------------- attention ---------------------------------

// per (node, head): alpha = <h[node, head, :], att[head, :]>
__global__ void k_alphas(const float* __restrict__ h,
                         const float* __restrict__ a_s,
                         const float* __restrict__ a_d,
                         float* __restrict__ as, float* __restrict__ ad,
                         int n, int heads, int fh, int stride) {
    int i = blockIdx.x * blockDim.x + threadIdx.x;
    if (i >= n * heads) return;
    int node = i / heads;
    int hd = i - node * heads;
    const float* hp = h + (long long)node * stride + hd * fh;
    const float* ws = a_s + hd * fh;
    const float* wd = a_d + hd * fh;
    float ss = 0.0f, sd = 0.0f;
    for (int f = 0; f < fh; ++f) {
        float v = hp[f];
        ss += v * ws[f];
        sd += v * wd[f];
    }
    as[i] = ss;
    ad[i] = sd;
}

__device__ __forceinline__ void atomicMaxF(float* addr, float val) {
    unsigned int* ua = (unsigned int*)addr;
    unsigned int old = *ua;
    while (__uint_as_float(old) < val) {
        unsigned int assumed = old;
        old = atomicCAS(ua, assumed, __float_as_uint(val));
        if (old == assumed) break;
    }
}

__global__ void k_edge_max(const int* __restrict__ src, const int* __restrict__ dst,
                           const float* __restrict__ as, const float* __restrict__ ad,
                           float* __restrict__ m, int E, int heads) {
    long long i = (long long)blockIdx.x * blockDim.x + threadIdx.x;
    if (i >= (long long)E * heads) return;
    int e = (int)(i / heads);
    int hd = (int)(i - (long long)e * heads);
    int s = src[e], d = dst[e];
    float ev = as[s * heads + hd] + ad[d * heads + hd];
    ev = fmaxf(ev, 0.2f * ev);                // leaky_relu(0.2)
    atomicMaxF(&m[d * heads + hd], ev);
}

__global__ void k_edge_exp(const int* __restrict__ src, const int* __restrict__ dst,
                           const float* __restrict__ as, const float* __restrict__ ad,
                           const float* __restrict__ m, float* __restrict__ p,
                           float* __restrict__ ssum, int E, int heads) {
    long long i = (long long)blockIdx.x * blockDim.x + threadIdx.x;
    if (i >= (long long)E * heads) return;
    int e = (int)(i / heads);
    int hd = (int)(i - (long long)e * heads);
    int s = src[e], d = dst[e];
    float ev = as[s * heads + hd] + ad[d * heads + hd];
    ev = fmaxf(ev, 0.2f * ev);
    float pv = __expf(ev - m[d * heads + hd]);
    p[i] = pv;
    atomicAdd(&ssum[d * heads + hd], pv);
}

// layer 1: 8 heads x 16 feats; 32 lanes per edge, 4 feats each (float4 gather)
__global__ void k_edge_agg1(const int* __restrict__ src, const int* __restrict__ dst,
                            const float* __restrict__ p, const float* __restrict__ ssum,
                            const float* __restrict__ h, float* __restrict__ agg, int E) {
    long long tid = (long long)blockIdx.x * blockDim.x + threadIdx.x;
    if (tid >= (long long)E * 32) return;
    int e = (int)(tid >> 5);
    int g = (int)(tid & 31);
    int f0 = g * 4;
    int hd = f0 >> 4;
    int s = src[e], d = dst[e];
    float w = p[(long long)e * 8 + hd] / fmaxf(ssum[d * 8 + hd], 1e-16f);
    const float4 hv = *(const float4*)(h + (long long)s * 128 + f0);
    float* o = agg + (long long)d * 128 + f0;
    atomicAdd(o + 0, w * hv.x);
    atomicAdd(o + 1, w * hv.y);
    atomicAdd(o + 2, w * hv.z);
    atomicAdd(o + 3, w * hv.w);
}

// layer 2: 1 head x 40 feats; 10 groups of 4 per edge
__global__ void k_edge_agg2(const int* __restrict__ src, const int* __restrict__ dst,
                            const float* __restrict__ p, const float* __restrict__ ssum,
                            const float* __restrict__ h, float* __restrict__ agg, int E) {
    long long tid = (long long)blockIdx.x * blockDim.x + threadIdx.x;
    if (tid >= (long long)E * 10) return;
    int e = (int)(tid / 10);
    int g = (int)(tid - (long long)e * 10);
    int f0 = g * 4;
    int s = src[e], d = dst[e];
    float w = p[e] / fmaxf(ssum[d], 1e-16f);
    const float4 hv = *(const float4*)(h + (long long)s * 40 + f0);
    float* o = agg + (long long)d * 40 + f0;
    atomicAdd(o + 0, w * hv.x);
    atomicAdd(o + 1, w * hv.y);
    atomicAdd(o + 2, w * hv.z);
    atomicAdd(o + 3, w * hv.w);
}

// out_bf16 = bf16( elu( agg + b1[i%128] ) )
__global__ void k_elu_bias_cvt(const float* __restrict__ agg,
                               const float* __restrict__ b,
                               unsigned short* __restrict__ out, long long n) {
    long long i = (long long)blockIdx.x * blockDim.x + threadIdx.x;
    if (i >= n) return;
    float v = agg[i] + b[i & 127];
    v = (v > 0.0f) ? v : expm1f(v);
    out[i] = f2bf(v);
}

__global__ void k_bias_out(const float* __restrict__ agg,
                           const float* __restrict__ b,
                           float* __restrict__ out, long long n) {
    long long i = (long long)blockIdx.x * blockDim.x + threadIdx.x;
    if (i >= n) return;
    out[i] = agg[i] + b[i % 40];
}

// ------------------------------- launch ------------------------------------

static inline int nblk(long long n, int t) { return (int)((n + t - 1) / t); }

extern "C" void kernel_launch(void* const* d_in, const int* in_sizes, int n_in,
                              void* d_out, int out_size, void* d_ws, size_t ws_size,
                              hipStream_t stream) {
    const float* x        = (const float*)d_in[0];
    const int*   src      = (const int*)d_in[1];
    const int*   dst      = (const int*)d_in[2];
    const float* W1       = (const float*)d_in[3];
    const float* att_s1   = (const float*)d_in[4];
    const float* att_d1   = (const float*)d_in[5];
    const float* b1       = (const float*)d_in[6];
    const float* W2       = (const float*)d_in[7];
    const float* att_s2   = (const float*)d_in[8];
    const float* att_d2   = (const float*)d_in[9];
    const float* b2       = (const float*)d_in[10];

    const int Nn = in_sizes[0] / 128;   // 100000
    const int Ee = in_sizes[1];         // 1600000

    // ---- workspace layout (256B-aligned slices) ----
    char* ws = (char*)d_ws;
    size_t off = 0;
    auto take = [&](size_t bytes) -> void* {
        void* pp = ws + off;
        off += (bytes + 255) & ~(size_t)255;
        return pp;
    };
    unsigned short* xbf  = (unsigned short*)take((size_t)Nn * 128 * 2);
    unsigned short* w1t  = (unsigned short*)take((size_t)128 * 128 * 2);
    unsigned short* w2t  = (unsigned short*)take((size_t)48 * 128 * 2);
    float* h1   = (float*)take((size_t)Nn * 128 * 4);
    float* as1  = (float*)take((size_t)Nn * 8 * 4);
    float* ad1  = (float*)take((size_t)Nn * 8 * 4);
    float* m1   = (float*)take((size_t)Nn * 8 * 4);
    float* s1   = (float*)take((size_t)Nn * 8 * 4);
    float* p1   = (float*)take((size_t)Ee * 8 * 4);
    float* agg1 = (float*)take((size_t)Nn * 128 * 4);
    unsigned short* x2bf = (unsigned short*)take((size_t)Nn * 128 * 2);
    float* h2   = (float*)take((size_t)Nn * 40 * 4);
    float* as2  = (float*)take((size_t)Nn * 4);
    float* ad2  = (float*)take((size_t)Nn * 4);
    float* m2   = (float*)take((size_t)Nn * 4);
    float* s2   = (float*)take((size_t)Nn * 4);
    float* p2   = (float*)take((size_t)Ee * 4);
    float* agg2 = (float*)take((size_t)Nn * 40 * 4);
    (void)ws_size;

    const int T = 256;
    const float NEG_INF = -INFINITY;

    // ---- init accumulators ----
    k_fill_f32<<<nblk((long long)Nn * 8, T), T, 0, stream>>>(m1, NEG_INF, (long long)Nn * 8);
    k_fill_f32<<<nblk((long long)Nn * 8, T), T, 0, stream>>>(s1, 0.0f, (long long)Nn * 8);
    k_fill_f32<<<nblk((long long)Nn * 128, T), T, 0, stream>>>(agg1, 0.0f, (long long)Nn * 128);
    k_fill_f32<<<nblk(Nn, T), T, 0, stream>>>(m2, NEG_INF, Nn);
    k_fill_f32<<<nblk(Nn, T), T, 0, stream>>>(s2, 0.0f, Nn);
    k_fill_f32<<<nblk((long long)Nn * 40, T), T, 0, stream>>>(agg2, 0.0f, (long long)Nn * 40);

    // ---- conversions ----
    k_cvt_bf16<<<nblk((long long)Nn * 128, T), T, 0, stream>>>(x, xbf, (long long)Nn * 128);
    k_transpose_cvt<<<nblk(128 * 128, T), T, 0, stream>>>(W1, w1t, 128, 128, 128);
    k_transpose_cvt<<<nblk(48 * 128, T), T, 0, stream>>>(W2, w2t, 128, 40, 48);

    // ---- layer 1 ----
    k_gemm_bf16_wmma<8><<<nblk(Nn, 128), T, 0, stream>>>(xbf, w1t, h1, Nn, 128);
    k_alphas<<<nblk((long long)Nn * 8, T), T, 0, stream>>>(h1, att_s1, att_d1, as1, ad1, Nn, 8, 16, 128);
    k_edge_max<<<nblk((long long)Ee * 8, T), T, 0, stream>>>(src, dst, as1, ad1, m1, Ee, 8);
    k_edge_exp<<<nblk((long long)Ee * 8, T), T, 0, stream>>>(src, dst, as1, ad1, m1, p1, s1, Ee, 8);
    k_edge_agg1<<<nblk((long long)Ee * 32, T), T, 0, stream>>>(src, dst, p1, s1, h1, agg1, Ee);
    k_elu_bias_cvt<<<nblk((long long)Nn * 128, T), T, 0, stream>>>(agg1, b1, x2bf, (long long)Nn * 128);

    // ---- layer 2 ----
    k_gemm_bf16_wmma<3><<<nblk(Nn, 128), T, 0, stream>>>(x2bf, w2t, h2, Nn, 40);
    k_alphas<<<nblk(Nn, T), T, 0, stream>>>(h2, att_s2, att_d2, as2, ad2, Nn, 1, 40, 40);
    k_edge_max<<<nblk(Ee, T), T, 0, stream>>>(src, dst, as2, ad2, m2, Ee, 1);
    k_edge_exp<<<nblk(Ee, T), T, 0, stream>>>(src, dst, as2, ad2, m2, p2, s2, Ee, 1);
    k_edge_agg2<<<nblk((long long)Ee * 10, T), T, 0, stream>>>(src, dst, p2, s2, h2, agg2, Ee);
    k_bias_out<<<nblk((long long)Nn * 40, T), T, 0, stream>>>(agg2, b2, (float*)d_out, (long long)Nn * 40);
}
